// BahdanauAttention_12171937317405
// MI455X (gfx1250) — matrix operand order repaired and checked by reading, not compiled, not run on hardware
//
#include <hip/hip_runtime.h>
#include <hip/hip_bf16.h>
#include <math.h>

// Problem constants (match reference)
#define BATCH   32
#define SEQ     2048
#define DIM     1024      // ENC_DIM == DEC_DIM
#define MT      64        // s-rows per workgroup (4 WMMA M-subtiles)
#define ASTRIDE 2064      // bytes per LDS A row: 1024*2 + 16 pad (bank-conflict-free, 16B aligned)

typedef __attribute__((ext_vector_type(8)))  float  v8f;
typedef __attribute__((ext_vector_type(16))) __bf16 v16bf;

union Frag16 { uint4 q[2]; v16bf v; };
union Pack4  { __bf16 h[4]; uint2 u; };

// Branch-free tanh: prefer the native CDNA5 V_TANH_F32 transcendental if the
// compiler exposes it; otherwise 2 trans ops (v_exp_f32 + v_rcp_f32), no EXEC
// divergence (ocml tanhf expands to a branchy polynomial path).
__device__ __forceinline__ float fast_tanh(float x) {
#if __has_builtin(__builtin_amdgcn_tanhf)
    return __builtin_amdgcn_tanhf(x);
#elif __has_builtin(__builtin_amdgcn_tanh_f32)
    return __builtin_amdgcn_tanh_f32(x);
#else
    float e = __expf(2.0f * x);                        // v_exp_f32 (inf for large x, 0 for very negative)
    return 1.0f - 2.0f * __builtin_amdgcn_rcpf(e + 1.0f);  // x>>0 -> 1, x<<0 -> -1
#endif
}

// ---------------------------------------------------------------------------
// 1) W_enc (K=1024, N=1024 f32 row-major) -> Wt bf16, N-major (Wt[n][k])
//    LDS-tiled transpose for coalescing.
__global__ void wenc_to_bf16_t_kernel(const float* __restrict__ W, __bf16* __restrict__ Wt) {
    __shared__ float tile[32][33];
    const int bx = blockIdx.x;   // k tile
    const int by = blockIdx.y;   // n tile
    const int tx = threadIdx.x;  // 0..31
    const int ty = threadIdx.y;  // 0..7
    #pragma unroll
    for (int i = 0; i < 4; ++i) {
        int k = bx * 32 + ty + i * 8;
        int n = by * 32 + tx;
        tile[ty + i * 8][tx] = W[k * DIM + n];
    }
    __syncthreads();
    #pragma unroll
    for (int i = 0; i < 4; ++i) {
        int n = by * 32 + ty + i * 8;
        int k = bx * 32 + tx;
        Wt[n * DIM + k] = (__bf16)tile[tx][ty + i * 8];
    }
}

// ---------------------------------------------------------------------------
// 2) dec_proj[b][d] = sum_k h[b][k] * W_dec[k][d]   (tiny GEMM, plain f32)
__global__ void decproj_kernel(const float* __restrict__ dh, const float* __restrict__ Wd,
                               float* __restrict__ dp) {
    __shared__ float h[DIM];
    const int b = blockIdx.y;
    const int d = blockIdx.x * 256 + threadIdx.x;
    for (int j = threadIdx.x; j < DIM; j += 256) h[j] = dh[b * DIM + j];
    __syncthreads();
    float acc = 0.f;
    #pragma unroll 4
    for (int k = 0; k < DIM; ++k) acc += h[k] * Wd[k * DIM + d];
    dp[b * DIM + d] = acc;
}

// ---------------------------------------------------------------------------
// 3) Main fused kernel: scores[b][s] = v . tanh(E[b,s,:] @ W_enc + dec_proj[b])
//    One WG = (b, 64-row s-tile). 8 waves, each owns a 128-wide N slice.
//    A tile staged in LDS as bf16; WMMA f32_16x16x32_bf16 over K.
__global__ void attn_scores_kernel(const float* __restrict__ enc,
                                   const __bf16* __restrict__ Wt,
                                   const float* __restrict__ dp,
                                   const float* __restrict__ vvec,
                                   const unsigned char* __restrict__ mask,
                                   float* __restrict__ scores) {
    extern __shared__ __align__(16) unsigned char smem[];
    unsigned char* aT  = smem;                                   // MT*ASTRIDE bytes, bf16 A tile
    float*         dpL = (float*)(smem + MT * ASTRIDE);          // 1024 f32
    float*         vL  = dpL + DIM;                              // 1024 f32
    float*         sL  = vL + DIM;                               // 64 f32 row scores

    const int b   = blockIdx.y;
    const int st  = blockIdx.x;          // 64-row tile index
    const int tid = threadIdx.x;         // 0..255

    // --- Stage A tile: 64 x 1024 f32 -> bf16 in LDS (padded row stride) ---
    const float4* esrc = (const float4*)(enc + ((size_t)b * SEQ + (size_t)st * MT) * DIM);
    #pragma unroll 4
    for (int j = 0; j < (MT * DIM / 4) / 256; ++j) {
        int i4 = tid + j * 256;
        int r  = i4 >> 8;            // row (DIM/4 = 256 float4 per row)
        int kq = i4 & 255;           // float4 index within row
        float4 f = esrc[r * 256 + kq];
        Pack4 pk;
        pk.h[0] = (__bf16)f.x; pk.h[1] = (__bf16)f.y;
        pk.h[2] = (__bf16)f.z; pk.h[3] = (__bf16)f.w;
        *(uint2*)(aT + r * ASTRIDE + kq * 8) = pk.u;
    }
    for (int j = tid; j < DIM; j += 256) { dpL[j] = dp[b * DIM + j]; vL[j] = vvec[j]; }
    if (tid < MT) sL[tid] = 0.f;
    __syncthreads();

    const int wave  = tid >> 5;
    const int lane  = tid & 31;
    const int half  = lane >> 4;     // 0: K-lo, 1: K-hi per ISA frag layout
    const int idx16 = lane & 15;     // A: row within subtile; B: column within tile
    const int klo   = half * 8;      // A fragment K sub-offset

    float part[4][8] = {};           // per-lane tanh-v partial per (m-subtile, acc vgpr)

    #pragma unroll 1
    for (int nt = 0; nt < 8; ++nt) {
        const int ncol = wave * 128 + nt * 16 + idx16;   // global N column this lane owns
        const unsigned char* brow = (const unsigned char*)(Wt + (size_t)ncol * DIM);

        // Prefetch next nt-tile's B row start into cache (global_prefetch_b8).
        if (nt < 7) __builtin_prefetch(brow + (size_t)16 * DIM * 2, 0, 0);

        v8f acc0 = {}, acc1 = {}, acc2 = {}, acc3 = {};
        #pragma unroll 4
        for (int kc = 0; kc < DIM / 32; ++kc) {
            // B fragment (32x16 bf16): lane = column; lane<16 -> K 0..7 & 16..23; lane>=16 -> K 8..15 & 24..31
            Frag16 bf;
            bf.q[0] = *(const uint4*)(brow + kc * 64 + half * 16);
            bf.q[1] = *(const uint4*)(brow + kc * 64 + half * 16 + 32);
            // A fragments (16x32 bf16): lane<16 -> K 0..7 & 16..23; lane>=16 -> K 8..15 & 24..31
            Frag16 a0, a1, a2, a3;
            {
                const unsigned char* ar = aT + (0 * 16 + idx16) * ASTRIDE + kc * 64 + klo * 2;
                a0.q[0] = *(const uint4*)ar; a0.q[1] = *(const uint4*)(ar + 32);
            }
            {
                const unsigned char* ar = aT + (1 * 16 + idx16) * ASTRIDE + kc * 64 + klo * 2;
                a1.q[0] = *(const uint4*)ar; a1.q[1] = *(const uint4*)(ar + 32);
            }
            {
                const unsigned char* ar = aT + (2 * 16 + idx16) * ASTRIDE + kc * 64 + klo * 2;
                a2.q[0] = *(const uint4*)ar; a2.q[1] = *(const uint4*)(ar + 32);
            }
            {
                const unsigned char* ar = aT + (3 * 16 + idx16) * ASTRIDE + kc * 64 + klo * 2;
                a3.q[0] = *(const uint4*)ar; a3.q[1] = *(const uint4*)(ar + 32);
            }
            acc0 = __builtin_amdgcn_wmma_f32_16x16x32_bf16(false, a0.v, false, bf.v, (short)0, acc0, false, false);
            acc1 = __builtin_amdgcn_wmma_f32_16x16x32_bf16(false, a1.v, false, bf.v, (short)0, acc1, false, false);
            acc2 = __builtin_amdgcn_wmma_f32_16x16x32_bf16(false, a2.v, false, bf.v, (short)0, acc2, false, false);
            acc3 = __builtin_amdgcn_wmma_f32_16x16x32_bf16(false, a3.v, false, bf.v, (short)0, acc3, false, false);
        }
        // acc element j: row = j + half*8 (within 16-row subtile), col = ncol
        const float dpv = dpL[ncol];
        const float vv  = vL[ncol];
        #pragma unroll
        for (int j = 0; j < 8; ++j) {
            part[0][j] += fast_tanh(acc0[j] + dpv) * vv;
            part[1][j] += fast_tanh(acc1[j] + dpv) * vv;
            part[2][j] += fast_tanh(acc2[j] + dpv) * vv;
            part[3][j] += fast_tanh(acc3[j] + dpv) * vv;
        }
    }

    // Reduce over the 16 columns each lane group covers, then across waves via LDS.
    #pragma unroll
    for (int m = 0; m < 4; ++m) {
        #pragma unroll
        for (int j = 0; j < 8; ++j) {
            float r = part[m][j];
            r += __shfl_xor(r, 1);
            r += __shfl_xor(r, 2);
            r += __shfl_xor(r, 4);
            r += __shfl_xor(r, 8);
            if (idx16 == 0) atomicAdd(&sL[m * 16 + half * 8 + j], r);
        }
    }
    __syncthreads();

    if (tid < MT) {
        int s = st * MT + tid;
        scores[b * SEQ + s] = mask[b * SEQ + s] ? -INFINITY : sL[tid];
    }
}

// ---------------------------------------------------------------------------
// 4) Row softmax over S=2048 (one WG per batch row)
__global__ void softmax_kernel(const float* __restrict__ scores, float* __restrict__ weights) {
    __shared__ float red[256];
    const int b = blockIdx.x, tid = threadIdx.x;
    const float* row = scores + b * SEQ;
    float mx = -INFINITY;
    for (int s = tid; s < SEQ; s += 256) mx = fmaxf(mx, row[s]);
    red[tid] = mx; __syncthreads();
    for (int o = 128; o > 0; o >>= 1) { if (tid < o) red[tid] = fmaxf(red[tid], red[tid + o]); __syncthreads(); }
    mx = red[0]; __syncthreads();
    float sum = 0.f;
    for (int s = tid; s < SEQ; s += 256) sum += __expf(row[s] - mx);
    red[tid] = sum; __syncthreads();
    for (int o = 128; o > 0; o >>= 1) { if (tid < o) red[tid] += red[tid + o]; __syncthreads(); }
    const float inv = 1.0f / red[0];
    for (int s = tid; s < SEQ; s += 256) weights[b * SEQ + s] = __expf(row[s] - mx) * inv;
}

// ---------------------------------------------------------------------------
// 5) context[b][e] = sum_s weights[b][s] * E[b][s][e]  (streaming, coalesced in e)
__global__ void context_kernel(const float* __restrict__ enc, const float* __restrict__ weights,
                               float* __restrict__ ctx) {
    __shared__ float w[SEQ];
    const int b = blockIdx.y;
    const int e = blockIdx.x * 256 + threadIdx.x;
    for (int s = threadIdx.x; s < SEQ; s += 256) w[s] = weights[b * SEQ + s];
    __syncthreads();
    const float* ep = enc + (size_t)b * SEQ * DIM + e;
    float acc = 0.f;
    #pragma unroll 4
    for (int s = 0; s < SEQ; ++s) acc += w[s] * ep[(size_t)s * DIM];
    ctx[b * DIM + e] = acc;
}

// ---------------------------------------------------------------------------
extern "C" void kernel_launch(void* const* d_in, const int* in_sizes, int n_in,
                              void* d_out, int out_size, void* d_ws, size_t ws_size,
                              hipStream_t stream) {
    const float*         dh   = (const float*)d_in[0];          // (B, DIM)
    const float*         enc  = (const float*)d_in[1];          // (B, S, DIM)
    const unsigned char* mask = (const unsigned char*)d_in[2];  // (B, S) bool
    const float*         Wenc = (const float*)d_in[3];          // (DIM, DIM)
    const float*         Wdec = (const float*)d_in[4];          // (DIM, DIM)
    const float*         v    = (const float*)d_in[5];          // (DIM,)

    float* out     = (float*)d_out;
    float* ctx     = out;                  // B*DIM   = 32768 floats (output 0)
    float* weights = out + BATCH * DIM;    // B*S     = 65536 floats (output 1)

    unsigned char* ws = (unsigned char*)d_ws;
    __bf16* Wt     = (__bf16*)ws;                                  // 2 MB: bf16 W_enc, N-major
    float*  dp     = (float*)(ws + (size_t)DIM * DIM * 2);         // 128 KB dec_proj
    float*  scores = (float*)(ws + (size_t)DIM * DIM * 2 + BATCH * DIM * 4); // 256 KB

    wenc_to_bf16_t_kernel<<<dim3(DIM / 32, DIM / 32), dim3(32, 8), 0, stream>>>(Wenc, Wt);
    decproj_kernel<<<dim3(DIM / 256, BATCH), 256, 0, stream>>>(dh, Wdec, dp);

    const size_t smem = (size_t)MT * ASTRIDE + DIM * 4 + DIM * 4 + MT * 4; // ~141 KB (< 320 KB/WGP)
    attn_scores_kernel<<<dim3(SEQ / MT, BATCH), 256, smem, stream>>>(enc, Wt, dp, v, mask, scores);

    softmax_kernel<<<BATCH, 256, 0, stream>>>(scores, weights);
    context_kernel<<<dim3(DIM / 256, BATCH), 256, 0, stream>>>(enc, weights, ctx);
}
